// AttentionWithFastKANTransform_16612933500962
// MI455X (gfx1250) — matrix operand order repaired and checked by reading, not compiled, not run on hardware
//
#include <hip/hip_runtime.h>

// ---------------------------------------------------------------------------
// Types for CDNA5 WMMA (wave32)
// ---------------------------------------------------------------------------
typedef __attribute__((ext_vector_type(16))) __bf16 v16bf;
typedef __attribute__((ext_vector_type(8)))  __bf16 v8bf;
typedef __attribute__((ext_vector_type(8)))  float  v8f;

#define NUM_GRIDS 8
#define DIMIN   512           // Q_DIM == K_DIM == V_DIM == TOT
#define AK      4608          // 512*8 basis + 512 silu
#define RTOT    4096          // B * L = 2 * 2048
#define SEQ     2048
#define NB      2
#define NH      8
#define HD      64

__device__ __forceinline__ v8f vzero8() {
  v8f z;
  #pragma unroll
  for (int i = 0; i < 8; ++i) z[i] = 0.0f;
  return z;
}
__device__ __forceinline__ v16bf cat8(v8bf lo, v8bf hi) {
  return __builtin_shufflevector(lo, hi, 0,1,2,3,4,5,6,7,8,9,10,11,12,13,14,15);
}
__device__ __forceinline__ float sigf(float x) { return 1.0f / (1.0f + __expf(-x)); }

// ---------------------------------------------------------------------------
// Pack spline weight [512,4096] and base weight [512,512] into one bf16
// matrix Wc[512, 4608] (row-major, K-contiguous) for a single fused GEMM.
// ---------------------------------------------------------------------------
__global__ __launch_bounds__(256)
void pack_weights_kernel(const float* __restrict__ sw, const float* __restrict__ bw,
                         __bf16* __restrict__ Wc) {
  size_t i = (size_t)blockIdx.x * 256 + threadIdx.x;   // over 512*4608
  int n = (int)(i / AK);
  int j = (int)(i % AK);
  float v = (j < 4096) ? sw[(size_t)n * 4096 + j] : bw[(size_t)n * DIMIN + (j - 4096)];
  Wc[i] = (__bf16)v;
}

// ---------------------------------------------------------------------------
// FastKAN activation: LayerNorm(row of 512) -> RBF basis (8 gaussians) and
// silu(x), emitted as one bf16 row of 4608 = [basis | silu].
// One block (256 threads, 8 waves) per row; each thread owns 2 columns.
// ---------------------------------------------------------------------------
__global__ __launch_bounds__(256)
void fastkan_act_kernel(const float* __restrict__ x,
                        const float* __restrict__ ln_s, const float* __restrict__ ln_b,
                        __bf16* __restrict__ Aout) {
  const int row  = blockIdx.x;
  const int tid  = threadIdx.x;
  const int wv   = tid >> 5;
  const int lane = tid & 31;
  const float* xr = x + (size_t)row * DIMIN;
  float x0 = xr[tid], x1 = xr[tid + 256];

  float s  = x0 + x1;
  float ss = x0 * x0 + x1 * x1;
  #pragma unroll
  for (int off = 1; off < 32; off <<= 1) {
    s  += __shfl_xor(s,  off, 32);
    ss += __shfl_xor(ss, off, 32);
  }
  __shared__ float red[16];
  if (lane == 0) { red[wv] = s; red[8 + wv] = ss; }
  __syncthreads();
  float ts = 0.f, tss = 0.f;
  #pragma unroll
  for (int i = 0; i < 8; ++i) { ts += red[i]; tss += red[8 + i]; }
  const float mu  = ts * (1.0f / DIMIN);
  const float var = tss * (1.0f / DIMIN) - mu * mu;
  const float inv = rsqrtf(var + 1e-5f);

  __bf16* arow = Aout + (size_t)row * AK;
  #pragma unroll
  for (int p = 0; p < 2; ++p) {
    const int   c  = tid + p * 256;
    const float xv = (p == 0) ? x0 : x1;
    const float xn = (xv - mu) * inv * ln_s[c] + ln_b[c];
    __bf16* dst = arow + c * NUM_GRIDS;
    #pragma unroll
    for (int g = 0; g < NUM_GRIDS; ++g) {
      const float gv = -2.0f + (float)g * (4.0f / 7.0f);
      const float t  = (xn - gv) * (7.0f / 4.0f);      // /DENOM, DENOM = 4/7
      dst[g] = (__bf16)__expf(-t * t);
    }
    arow[4096 + c] = (__bf16)(xv * sigf(xv));          // silu base path
  }
}

// ---------------------------------------------------------------------------
// WMMA GEMM: C[R,512] = A[R,4608](bf16) @ Wc[512,4608]^T + bias, *scale.
// Block = 256 thr (8 waves) computing a 128x128 tile; wave = 64Mx32N (4x2
// subtiles of 16x16). Fragments use contiguous k order (k = 16*half + s) in both A and B.
// Epilogue can write fp32, bf16, and/or head-transposed bf16 (V^T for attn).
// ---------------------------------------------------------------------------
__global__ __launch_bounds__(256)
void wmma_gemm_kernel(const __bf16* __restrict__ A, const __bf16* __restrict__ W,
                      const float* __restrict__ bias, float scale,
                      float* __restrict__ outF, __bf16* __restrict__ outB,
                      __bf16* __restrict__ outBT) {
  const int tid  = threadIdx.x;
  const int wv   = tid >> 5;
  const int lane = tid & 31;
  const int hlf  = lane >> 4;       // lane half selects K sub-ranges
  const int lr   = lane & 15;

  const int m0 = blockIdx.x * 128 + (wv >> 2) * 64;   // 4 M-subtiles of 16
  const int n0 = blockIdx.y * 128 + (wv & 3) * 32;    // 2 N-subtiles of 16

  v8f acc[4][2];
  #pragma unroll
  for (int mt = 0; mt < 4; ++mt)
    #pragma unroll
    for (int nt = 0; nt < 2; ++nt) acc[mt][nt] = vzero8();

  for (int k0 = 0; k0 < AK; k0 += 32) {
    if (k0 + 128 < AK)
      __builtin_prefetch(W + (size_t)(n0 + lr) * AK + k0 + 128, 0, 1);

    // B fragments: lane half 0 -> K k0..k0+15, half 1 -> K k0+16..k0+31
    v16bf bfr[2];
    #pragma unroll
    for (int nt = 0; nt < 2; ++nt) {
      const int n = n0 + nt * 16 + lr;
      bfr[nt] = *reinterpret_cast<const v16bf*>(W + (size_t)n * AK + k0 + hlf * 16);
    }
    // A fragments: elems 0..15 = K k0+hlf*16.. (contiguous, matches B)
    v16bf afr[4];
    #pragma unroll
    for (int mt = 0; mt < 4; ++mt) {
      const int m = m0 + mt * 16 + lr;
      const __bf16* ap = A + (size_t)m * AK + k0 + hlf * 16;
      afr[mt] = cat8(*reinterpret_cast<const v8bf*>(ap),
                     *reinterpret_cast<const v8bf*>(ap + 8));
    }
    #pragma unroll
    for (int mt = 0; mt < 4; ++mt)
      #pragma unroll
      for (int nt = 0; nt < 2; ++nt)
        acc[mt][nt] = __builtin_amdgcn_wmma_f32_16x16x32_bf16(
            false, afr[mt], false, bfr[nt], (short)0, acc[mt][nt], false, false);
  }

  #pragma unroll
  for (int mt = 0; mt < 4; ++mt) {
    #pragma unroll
    for (int nt = 0; nt < 2; ++nt) {
      const int   n = n0 + nt * 16 + lr;
      const float b = bias ? bias[n] : 0.0f;
      #pragma unroll
      for (int r = 0; r < 8; ++r) {
        const int   m = m0 + mt * 16 + hlf * 8 + r;  // C layout: half picks rows 0..7 / 8..15
        const float v = (acc[mt][nt][r] + b) * scale;
        if (outF) outF[(size_t)m * DIMIN + n] = v;
        if (outB) outB[(size_t)m * DIMIN + n] = (__bf16)v;
        if (outBT) {  // store V^T as [b, h, d, seq] for contiguous A-frags in attention
          const int bb = m >> 11, l = m & (SEQ - 1);
          const int hh = n >> 6,  d = n & (HD - 1);
          outBT[(((size_t)(bb * NH + hh) * HD + d) << 11) + l] = (__bf16)v;
        }
      }
    }
  }
}

// ---------------------------------------------------------------------------
// Flash attention, one wave per 16-query tile (4 waves / block).
// S^T = K·Q^T  (softmax over k lives inside each lane's accumulator rows;
// the two lanes sharing a q column exchange stats via shfl_xor 16),
// then O^T = V^T·P^T accumulated online.  8 WMMAs per 32 keys.
// ---------------------------------------------------------------------------
__global__ __launch_bounds__(128)
void flash_attn_kernel(const __bf16* __restrict__ wq, const __bf16* __restrict__ wk,
                       const __bf16* __restrict__ wvT, float* __restrict__ outO) {
  const int bh = blockIdx.x;          // B*H = 16
  const int b  = bh >> 3, h = bh & 7;
  const int wv   = threadIdx.x >> 5;
  const int lane = threadIdx.x & 31;
  const int hlf  = lane >> 4;
  const int lr   = lane & 15;
  const int qtile = blockIdx.y * 64 + wv * 16;

  const __bf16* qbase = wq + (size_t)(b * SEQ) * DIMIN + h * HD;
  const __bf16* kbase = wk + (size_t)(b * SEQ) * DIMIN + h * HD;
  const __bf16* vtb   = wvT + (size_t)(b * NH + h) * HD * SEQ;

  // Q^T B-fragments (fixed for the whole k loop): d-chunks [0,32) and [32,64)
  v16bf bq0, bq1;
  {
    const __bf16* qp = qbase + (size_t)(qtile + lr) * DIMIN;
    bq0 = *reinterpret_cast<const v16bf*>(qp + hlf * 16);
    bq1 = *reinterpret_cast<const v16bf*>(qp + 32 + hlf * 16);
  }

  float m = -1e30f, l = 0.0f;
  v8f acc[4];
  #pragma unroll
  for (int f = 0; f < 4; ++f) acc[f] = vzero8();

  for (int k0 = 0; k0 < SEQ; k0 += 32) {
    // S^T fragments for k rows [k0,k0+16) and [k0+16,k0+32), accumulated over d
    v8f s[2];
    #pragma unroll
    for (int kk = 0; kk < 2; ++kk) {
      s[kk] = vzero8();
      const __bf16* kp = kbase + (size_t)(k0 + kk * 16 + lr) * DIMIN;
      v16bf a0 = cat8(*reinterpret_cast<const v8bf*>(kp + hlf * 16),
                      *reinterpret_cast<const v8bf*>(kp + hlf * 16 + 8));
      v16bf a1 = cat8(*reinterpret_cast<const v8bf*>(kp + 32 + hlf * 16),
                      *reinterpret_cast<const v8bf*>(kp + 32 + hlf * 16 + 8));
      s[kk] = __builtin_amdgcn_wmma_f32_16x16x32_bf16(false, a0, false, bq0, (short)0, s[kk], false, false);
      s[kk] = __builtin_amdgcn_wmma_f32_16x16x32_bf16(false, a1, false, bq1, (short)0, s[kk], false, false);
    }

    // ---- online softmax over k (column q = lr is shared by lanes lr, lr+16)
    float tmax = -1e30f;
    #pragma unroll
    for (int r = 0; r < 8; ++r) tmax = fmaxf(tmax, fmaxf(s[0][r], s[1][r]));
    tmax = fmaxf(tmax, __shfl_xor(tmax, 16, 32));
    const float mnew = fmaxf(m, tmax);
    const float corr = __expf(m - mnew);
    float p0[8], p1[8], ls = 0.0f;
    #pragma unroll
    for (int r = 0; r < 8; ++r) {
      p0[r] = __expf(s[0][r] - mnew); ls += p0[r];
      p1[r] = __expf(s[1][r] - mnew); ls += p1[r];
    }
    ls += __shfl_xor(ls, 16, 32);
    l = l * corr + ls;
    m = mnew;
    #pragma unroll
    for (int f = 0; f < 4; ++f)
      #pragma unroll
      for (int r = 0; r < 8; ++r) acc[f][r] *= corr;

    // ---- build P^T B-fragment (K rows k0..k0+31, cols q) from lane pairs
    float q0[8], q1[8];
    #pragma unroll
    for (int r = 0; r < 8; ++r) {
      q0[r] = __shfl_xor(p0[r], 16, 32);
      q1[r] = __shfl_xor(p1[r], 16, 32);
    }
    v16bf bp;
    #pragma unroll
    for (int r = 0; r < 8; ++r) {
      bp[r]     = (__bf16)(hlf ? q1[r] : p0[r]);   // K rel rows 0..7 / 16..23
      bp[r + 8] = (__bf16)(hlf ? p1[r] : q0[r]);   // K rel rows 8..15 / 24..31
    }

    // ---- O^T += V^T(16d x 32k) @ P^T(32k x 16q), 4 d-subtiles
    #pragma unroll
    for (int f = 0; f < 4; ++f) {
      const __bf16* vp = vtb + (size_t)(f * 16 + lr) * SEQ + k0 + hlf * 16;
      v16bf av = cat8(*reinterpret_cast<const v8bf*>(vp),
                      *reinterpret_cast<const v8bf*>(vp + 8));
      acc[f] = __builtin_amdgcn_wmma_f32_16x16x32_bf16(false, av, false, bp, (short)0, acc[f], false, false);
    }
  }

  const float invl = 1.0f / l;
  const int   q    = qtile + lr;
  float* orow = outO + (size_t)(b * SEQ + q) * DIMIN + h * HD;
  #pragma unroll
  for (int f = 0; f < 4; ++f)
    #pragma unroll
    for (int r = 0; r < 8; ++r)
      orow[f * 16 + hlf * 8 + r] = acc[f][r] * invl;
}

// ---------------------------------------------------------------------------
// o = sigmoid(g) * o   (in place on the attention output buffer)
// ---------------------------------------------------------------------------
__global__ __launch_bounds__(256)
void gate_kernel(float* __restrict__ o, const float* __restrict__ g) {
  size_t i = (size_t)blockIdx.x * 256 + threadIdx.x;
  o[i] = o[i] * sigf(g[i]);
}

// ---------------------------------------------------------------------------
extern "C" void kernel_launch(void* const* d_in, const int* in_sizes, int n_in,
                              void* d_out, int out_size, void* d_ws, size_t ws_size,
                              hipStream_t stream) {
  const float* q = (const float*)d_in[0];
  const float* k = (const float*)d_in[1];
  const float* v = (const float*)d_in[2];
  // Parameter base indices: lq=3, lk=8, lv=13, lo=18, lg=23  (ln_s,ln_b,sw,bw,bb)

  char* w = (char*)d_ws;
  __bf16* Aact = (__bf16*)w;  w += (size_t)RTOT * AK * 2;       // 37.75 MB
  __bf16* Wc   = (__bf16*)w;  w += (size_t)DIMIN * AK * 2;      //  4.72 MB
  __bf16* wqb  = (__bf16*)w;  w += (size_t)RTOT * DIMIN * 2;    //  4 MB
  __bf16* wkb  = (__bf16*)w;  w += (size_t)RTOT * DIMIN * 2;    //  4 MB
  __bf16* wvT  = (__bf16*)w;  w += (size_t)RTOT * DIMIN * 2;    //  4 MB
  float*  oat  = (float*)w;   w += (size_t)RTOT * DIMIN * 4;    //  8 MB
  float*  gbuf = (float*)w;   w += (size_t)RTOT * DIMIN * 4;    //  8 MB

  const dim3 gemmGrid(RTOT / 128, DIMIN / 128);
  const int packBlocks = (DIMIN * AK) / 256;

  auto run_fastkan = [&](const float* x, int p, float scale,
                         float* oF, __bf16* oB, __bf16* oBT) {
    pack_weights_kernel<<<packBlocks, 256, 0, stream>>>(
        (const float*)d_in[p + 2], (const float*)d_in[p + 3], Wc);
    fastkan_act_kernel<<<RTOT, 256, 0, stream>>>(
        x, (const float*)d_in[p], (const float*)d_in[p + 1], Aact);
    wmma_gemm_kernel<<<gemmGrid, 256, 0, stream>>>(
        Aact, Wc, (const float*)d_in[p + 4], scale, oF, oB, oBT);
  };

  run_fastkan(q, 3,  0.125f, nullptr, wqb, nullptr);   // wq * D^-0.5
  run_fastkan(k, 8,  1.0f,   nullptr, wkb, nullptr);   // wk
  run_fastkan(v, 13, 1.0f,   nullptr, nullptr, wvT);   // wv stored transposed
  run_fastkan(q, 23, 1.0f,   gbuf,    nullptr, nullptr); // gate pre-activation

  flash_attn_kernel<<<dim3(NB * NH, SEQ / 64), 128, 0, stream>>>(wqb, wkb, wvT, oat);
  gate_kernel<<<(RTOT * DIMIN) / 256, 256, 0, stream>>>(oat, gbuf);

  run_fastkan(oat, 18, 1.0f, (float*)d_out, nullptr, nullptr); // final FastKAN
}
